// GNN_Transformer_56178172231999
// MI455X (gfx1250) — compile-verified, matrix operand order checked by reading, not run on hardware
//
#include <hip/hip_runtime.h>
#include <hip/hip_bf16.h>

// ---------------- problem constants ----------------
constexpr int Bg    = 256;            // graphs
constexpr int NODES = 90;
constexpr int Nn    = Bg * NODES;     // 23040 nodes per modality
constexpr int Hd    = 128;            // hidden
constexpr int HEADS = 4;
constexpr int DHd   = Hd / HEADS;     // 32
constexpr int DFF   = 512;
constexpr int Ed    = 460800;         // edges per modality
constexpr int Sq    = 2 * NODES;      // 180 tokens per graph in transformer
constexpr int NT    = Bg * Sq;        // 46080 transformer rows
constexpr int KPAD  = 96;             // embedding K=90 padded to 96
constexpr float BN_EPS = 1e-5f;

typedef __attribute__((ext_vector_type(16))) _Float16 v16h;
typedef __attribute__((ext_vector_type(8)))  _Float16 v8h;
typedef __attribute__((ext_vector_type(4)))  _Float16 v4h;
typedef __attribute__((ext_vector_type(8)))  float    v8f;

static inline int cdiv(int a, int b) { return (a + b - 1) / b; }

// ---------------- WMMA GEMM (f16 in, f32 acc): C = act(A[rows x K] @ W[M x K]^T + bias)
// block = (32,4): one wave per 16x16 output tile, 4 col-tiles per block.
// grid  = (rows/16, M/64). Requires rows%16==0, M%64==0, K%32==0.
// If C16 != nullptr the result is stored as f16 into C16 instead of f32 into C.
__global__ void wmma_gemm_f16_kernel(const _Float16* __restrict__ A,
                                     const _Float16* __restrict__ W,
                                     const float* __restrict__ bias,
                                     float* __restrict__ C,
                                     _Float16* __restrict__ C16,
                                     int K, int M, int relu) {
  const int lane = threadIdx.x;                        // 0..31
  const int rowT = blockIdx.x * 16;
  const int colT = (blockIdx.y * 4 + threadIdx.y) * 16;
  const int m    = lane & 15;                          // A-row / B-col / D-col selector
  const int kh   = lane >> 4;                          // K-half selector
  const _Float16* __restrict__ Arow = A + (size_t)(rowT + m) * K + (kh << 3);
  const _Float16* __restrict__ Wrow = W + (size_t)(colT + m) * K + (kh << 3);

  v8f acc = {};
  for (int k0 = 0; k0 < K; k0 += 32) {
    // A fragment: halves K = k0 + kh*8 + [0..8)  and  k0 + 16 + kh*8 + [0..8)
    const v8h a0 = *(const v8h*)(Arow + k0);
    const v8h a1 = *(const v8h*)(Arow + k0 + 16);
    const v8h b0 = *(const v8h*)(Wrow + k0);
    const v8h b1 = *(const v8h*)(Wrow + k0 + 16);
    const v16h af = __builtin_shufflevector(a0, a1, 0,1,2,3,4,5,6,7,8,9,10,11,12,13,14,15);
    const v16h bf = __builtin_shufflevector(b0, b1, 0,1,2,3,4,5,6,7,8,9,10,11,12,13,14,15);
    acc = __builtin_amdgcn_wmma_f32_16x16x32_f16(false, af, false, bf,
                                                 (short)0, acc, false, false);
  }
  // D layout: lane -> col (lane&15), VGPR v -> row v + (lane>>4)*8
  const int col = colT + m;
  const float bv = bias ? bias[col] : 0.0f;
#pragma unroll
  for (int v = 0; v < 8; ++v) {
    const int row = rowT + (kh << 3) + v;
    float val = acc[v] + bv;
    if (relu) val = fmaxf(val, 0.0f);
    if (C16) C16[(size_t)row * M + col] = (_Float16)val;
    else     C[(size_t)row * M + col] = val;
  }
}

// ---------------- conversion kernels ----------------
// f32 -> f16, 4 elements per thread (n must be a multiple of 4)
__global__ void cvt4_f16_kernel(const float* __restrict__ x,
                                _Float16* __restrict__ y, int n4) {
  int i = blockIdx.x * blockDim.x + threadIdx.x;
  if (i >= n4) return;
  const float4 v = ((const float4*)x)[i];
  v4h o;
  o[0] = (_Float16)v.x; o[1] = (_Float16)v.y;
  o[2] = (_Float16)v.z; o[3] = (_Float16)v.w;
  ((v4h*)y)[i] = o;
}

// f32 [rows x Kin] -> f16 [rows x Kout], zero padded (Kout > Kin)
__global__ void pad_cvt_f16_kernel(const float* __restrict__ x,
                                   _Float16* __restrict__ y,
                                   int rows, int Kin, int Kout) {
  int i = blockIdx.x * blockDim.x + threadIdx.x;
  if (i >= rows * Kout) return;
  const int r = i / Kout;
  const int c = i - r * Kout;
  y[i] = (c < Kin) ? (_Float16)x[(size_t)r * Kin + c] : (_Float16)0.0f;
}

// ---------------- elementwise / graph kernels ----------------
__global__ void zero_kernel(float* p, int n) {
  int i = blockIdx.x * blockDim.x + threadIdx.x;
  if (i < n) p[i] = 0.0f;
}

__global__ void edge_scatter_kernel(const int* __restrict__ ei,
                                    const float* __restrict__ x,
                                    float* __restrict__ agg, int nE) {
  int i = blockIdx.x * blockDim.x + threadIdx.x;
  if (i >= nE * Hd) return;
  const int e = i >> 7;
  const int f = i & (Hd - 1);
  const int src = ei[e];
  const int dst = ei[nE + e];
  atomicAdd(&agg[(size_t)dst * Hd + f], x[(size_t)src * Hd + f]);
}

__global__ void gin_combine_kernel(const float* __restrict__ x,
                                   float* __restrict__ pre,
                                   const float* __restrict__ epsArr,
                                   int layer, int n) {
  int i = blockIdx.x * blockDim.x + threadIdx.x;
  if (i >= n) return;
  const float e = 1.0f + epsArr[layer];
  pre[i] = e * x[i] + pre[i];
}

// BatchNorm stats over rows (biased variance). One block (256 thr) per column.
__global__ void bn_stats_kernel(const float* __restrict__ x, int rows,
                                float* __restrict__ mean, float* __restrict__ var) {
  const int c = blockIdx.x;
  const int t = threadIdx.x;
  float s = 0.0f, s2 = 0.0f;
  for (int r = t; r < rows; r += 256) {
    const float v = x[(size_t)r * Hd + c];
    s += v; s2 += v * v;
  }
  __shared__ float rs[256], rs2[256];
  rs[t] = s; rs2[t] = s2;
  __syncthreads();
  for (int off = 128; off > 0; off >>= 1) {
    if (t < off) { rs[t] += rs[t + off]; rs2[t] += rs2[t + off]; }
    __syncthreads();
  }
  if (t == 0) {
    const float m = rs[0] / (float)rows;
    mean[c] = m;
    var[c]  = rs2[0] / (float)rows - m * m;
  }
}

__global__ void bn_apply_relu_kernel(float* __restrict__ x, int n,
                                     const float* __restrict__ mean,
                                     const float* __restrict__ var,
                                     const float* __restrict__ g,
                                     const float* __restrict__ b) {
  int i = blockIdx.x * blockDim.x + threadIdx.x;
  if (i >= n) return;
  const int c = i & (Hd - 1);
  const float v = (x[i] - mean[c]) * rsqrtf(var[c] + BN_EPS) * g[c] + b[c];
  x[i] = fmaxf(v, 0.0f);
}

// out = LayerNorm(x + r) ; one block of 128 threads per row
__global__ void add_ln_kernel(const float* __restrict__ x,
                              const float* __restrict__ r,
                              const float* __restrict__ g,
                              const float* __restrict__ b,
                              float* __restrict__ out) {
  const int row = blockIdx.x;
  const int t   = threadIdx.x;
  __shared__ float red[Hd];
  __shared__ float s_mean, s_rstd;
  const float v = x[(size_t)row * Hd + t] + r[(size_t)row * Hd + t];
  red[t] = v;
  __syncthreads();
  for (int off = 64; off > 0; off >>= 1) {
    if (t < off) red[t] += red[t + off];
    __syncthreads();
  }
  if (t == 0) s_mean = red[0] * (1.0f / Hd);
  __syncthreads();
  const float d = v - s_mean;
  red[t] = d * d;
  __syncthreads();
  for (int off = 64; off > 0; off >>= 1) {
    if (t < off) red[t] += red[t + off];
    __syncthreads();
  }
  if (t == 0) s_rstd = rsqrtf(red[0] * (1.0f / Hd) + BN_EPS);
  __syncthreads();
  out[(size_t)row * Hd + t] = d * s_rstd * g[t] + b[t];
}

// attention: one block (256 thr) per (b*HEADS+h, q). qkv: [NT x 3H], o: [NT x H]
__global__ void attn_kernel(const float* __restrict__ qkv, float* __restrict__ o) {
  const int bh = blockIdx.x;
  const int b  = bh / HEADS;
  const int h  = bh - b * HEADS;
  const int q  = blockIdx.y;
  const int j  = threadIdx.x;
  const float scale = 0.17677669529663687f;   // 1/sqrt(32)
  __shared__ float qv[DHd];
  __shared__ float aw[256];
  __shared__ float red[256];
  __shared__ float s_max, s_inv;
  const size_t rowq = (size_t)(b * Sq + q) * (3 * Hd);
  if (j < DHd) qv[j] = qkv[rowq + h * DHd + j];
  __syncthreads();
  float score = -INFINITY;
  if (j < Sq) {
    const float* kj = qkv + (size_t)(b * Sq + j) * (3 * Hd) + Hd + h * DHd;
    float acc = 0.0f;
#pragma unroll
    for (int d = 0; d < DHd; ++d) acc += qv[d] * kj[d];
    score = acc * scale;
  }
  red[j] = score;
  __syncthreads();
  for (int off = 128; off > 0; off >>= 1) {
    if (j < off) red[j] = fmaxf(red[j], red[j + off]);
    __syncthreads();
  }
  if (j == 0) s_max = red[0];
  __syncthreads();
  const float e = (j < Sq) ? __expf(score - s_max) : 0.0f;
  aw[j] = e;
  red[j] = e;
  __syncthreads();
  for (int off = 128; off > 0; off >>= 1) {
    if (j < off) red[j] += red[j + off];
    __syncthreads();
  }
  if (j == 0) s_inv = 1.0f / red[0];
  __syncthreads();
  if (j < DHd) {
    float acc = 0.0f;
    for (int jj = 0; jj < Sq; ++jj)
      acc += aw[jj] * qkv[(size_t)(b * Sq + jj) * (3 * Hd) + 2 * Hd + h * DHd + j];
    o[(size_t)(b * Sq + q) * Hd + h * DHd + j] = acc * s_inv;
  }
}

// fuse tokens: row b*180+s : s<90 -> sc node, s>=90 -> fc node
__global__ void concat_kernel(const float* __restrict__ sc,
                              const float* __restrict__ fc,
                              float* __restrict__ out, int n) {
  int i = blockIdx.x * blockDim.x + threadIdx.x;
  if (i >= n) return;
  const int f = i & (Hd - 1);
  const int r = i >> 7;
  const int b = r / Sq;
  const int s = r - b * Sq;
  out[i] = (s < NODES) ? sc[((size_t)b * NODES + s) * Hd + f]
                       : fc[((size_t)b * NODES + (s - NODES)) * Hd + f];
}

// extract the fc half of the fused transformer output
__global__ void extract_kernel(const float* __restrict__ fus,
                               float* __restrict__ out, int n) {
  int i = blockIdx.x * blockDim.x + threadIdx.x;
  if (i >= n) return;
  const int f = i & (Hd - 1);
  const int r = i >> 7;
  const int b = r / NODES;
  const int s = r - b * NODES;
  out[i] = fus[((size_t)b * Sq + NODES + s) * Hd + f];
}

__global__ void pool_kernel(const float* __restrict__ x, float* __restrict__ pooled) {
  const int b = blockIdx.x;
  const int t = threadIdx.x;
  float acc = 0.0f;
  for (int s = 0; s < NODES; ++s) acc += x[((size_t)b * NODES + s) * Hd + t];
  pooled[(size_t)b * Hd + t] = acc * (1.0f / (float)NODES);
}

// logit = 0.5*(clf(pooled_sc)+clf(pooled_fc)); one block (128 thr) per graph
__global__ void classifier_kernel(const float* __restrict__ psc,
                                  const float* __restrict__ pfc,
                                  const float* __restrict__ w1,
                                  const float* __restrict__ b1,
                                  const float* __restrict__ w2,
                                  const float* __restrict__ b2,
                                  float* __restrict__ out) {
  const int b = blockIdx.x;
  const int t = threadIdx.x;
  __shared__ float h1s[Hd];
  __shared__ float lg[2][2];
  for (int which = 0; which < 2; ++which) {
    const float* p = (which == 0 ? psc : pfc) + (size_t)b * Hd;
    float acc = b1[t];
    for (int k = 0; k < Hd; ++k) acc += p[k] * w1[t * Hd + k];
    h1s[t] = fmaxf(acc, 0.0f);
    __syncthreads();
    if (t < 2) {
      float a = b2[t];
      for (int k = 0; k < Hd; ++k) a += h1s[k] * w2[t * Hd + k];
      lg[which][t] = a;
    }
    __syncthreads();
  }
  if (t < 2) out[b * 2 + t] = 0.5f * (lg[0][t] + lg[1][t]);
}

// ---------------- host orchestration ----------------
extern "C" void kernel_launch(void* const* d_in, const int* in_sizes, int n_in,
                              void* d_out, int out_size, void* d_ws, size_t ws_size,
                              hipStream_t stream) {
  (void)in_sizes; (void)n_in; (void)out_size; (void)ws_size;
  // inputs (setup_inputs dict leaf order)
  const float* x_sc  = (const float*)d_in[0];
  const float* x_fc  = (const float*)d_in[1];
  const int*   sc_ei = (const int*)d_in[2];
  const int*   fc_ei = (const int*)d_in[3];
  const float* emb_sc_w = (const float*)d_in[4];
  const float* emb_sc_b = (const float*)d_in[5];
  const float* emb_fc_w = (const float*)d_in[6];
  const float* emb_fc_b = (const float*)d_in[7];
  // gin_sc: eps,w1,b1,g1,be1,w2,b2 -> 8..14 ; gin_fc -> 15..21
  const float* gsc[7]; const float* gfc[7];
  for (int k = 0; k < 7; ++k) { gsc[k] = (const float*)d_in[8 + k]; gfc[k] = (const float*)d_in[15 + k]; }
  // tf: qkv_w,qkv_b,out_w,out_b,ff1_w,ff1_b,ff2_w,ff2_b,ln1_g,ln1_b,ln2_g,ln2_b -> 22..33
  const float* tf_qkv_w = (const float*)d_in[22];
  const float* tf_qkv_b = (const float*)d_in[23];
  const float* tf_out_w = (const float*)d_in[24];
  const float* tf_out_b = (const float*)d_in[25];
  const float* tf_ff1_w = (const float*)d_in[26];
  const float* tf_ff1_b = (const float*)d_in[27];
  const float* tf_ff2_w = (const float*)d_in[28];
  const float* tf_ff2_b = (const float*)d_in[29];
  const float* tf_ln1_g = (const float*)d_in[30];
  const float* tf_ln1_b = (const float*)d_in[31];
  const float* tf_ln2_g = (const float*)d_in[32];
  const float* tf_ln2_b = (const float*)d_in[33];
  const float* bns_g = (const float*)d_in[34];
  const float* bns_b = (const float*)d_in[35];
  const float* cls_w1 = (const float*)d_in[36];
  const float* cls_b1 = (const float*)d_in[37];
  const float* cls_w2 = (const float*)d_in[38];
  const float* cls_b2 = (const float*)d_in[39];

  // workspace carve-up
  const size_t NH = (size_t)Nn * Hd;     // 2,949,120
  const size_t TH = (size_t)NT * Hd;     // 5,898,240
  float* ws   = (float*)d_ws;
  float* sc   = ws;                       // N x H
  float* fcb  = sc + NH;                  // N x H
  float* pre  = fcb + NH;                 // N x H
  float* h1   = pre + NH;                 // N x H
  float* fus  = h1 + NH;                  // NT x H
  float* x1   = fus + TH;                 // NT x H
  float* ao   = x1 + TH;                  // NT x H
  float* qkv  = ao + TH;                  // NT x 3H (f32, read by attention)
  float* mean_ = qkv + (size_t)NT * 3 * Hd;
  float* var_  = mean_ + Hd;
  float* psc   = var_ + Hd;               // B x H
  float* pfc   = psc + (size_t)Bg * Hd;   // B x H
  // f16 scratch regions (2-byte elems), placed after the f32 regions
  _Float16* A16 = (_Float16*)(pfc + (size_t)Bg * Hd);   // max(NT*Hd, Nn*KPAD) halves
  _Float16* H16 = A16 + TH;                             // NT x DFF halves (ff1 out)
  _Float16* W16 = H16 + (size_t)NT * DFF;               // max M*K = 65536 halves

  float* out    = (float*)d_out;
  float* sc_out = out + (size_t)Bg * 2;          // final sc (N x H)
  float* fc_out = sc_out + NH;                   // final fc (N x H)

  const dim3 gblk(32, 4);
  auto ew = [&](int n) { return dim3((unsigned)cdiv(n, 256)); };
  auto cvt = [&](const float* src, _Float16* dst, int n) {
    cvt4_f16_kernel<<<ew(n / 4), 256, 0, stream>>>(src, dst, n / 4);
  };
  // full GEMM from f32 operands: converts A and W to f16 first
  auto gemm = [&](const float* A, const float* W, const float* bias, float* C,
                  int rows, int K, int M, int relu, _Float16* C16 = nullptr) {
    cvt(A, A16, rows * K);
    cvt(W, W16, M * K);
    dim3 grd((unsigned)(rows / 16), (unsigned)(M / 64));
    wmma_gemm_f16_kernel<<<grd, gblk, 0, stream>>>(A16, W16, bias, C, C16, K, M, relu);
  };
  // GEMM whose A operand is already f16 (ff2 consuming ff1's output)
  auto gemm_a16 = [&](const _Float16* A, const float* W, const float* bias, float* C,
                      int rows, int K, int M, int relu) {
    cvt(W, W16, M * K);
    dim3 grd((unsigned)(rows / 16), (unsigned)(M / 64));
    wmma_gemm_f16_kernel<<<grd, gblk, 0, stream>>>(A, W16, bias, C, nullptr, K, M, relu);
  };
  // embedding GEMM: K=90 padded to 96 on both operands
  auto gemm_emb = [&](const float* X, const float* W, const float* bias, float* C) {
    pad_cvt_f16_kernel<<<ew(Nn * KPAD), 256, 0, stream>>>(X, A16, Nn, NODES, KPAD);
    pad_cvt_f16_kernel<<<ew(Hd * KPAD), 256, 0, stream>>>(W, W16, Hd, NODES, KPAD);
    dim3 grd((unsigned)(Nn / 16), (unsigned)(Hd / 64));
    wmma_gemm_f16_kernel<<<grd, gblk, 0, stream>>>(A16, W16, bias, C, nullptr, KPAD, Hd, 0);
  };

  auto gin = [&](const float* xin, float* xout, const int* ei, const float* const* p, int layer) {
    zero_kernel<<<ew((int)NH), 256, 0, stream>>>(pre, (int)NH);
    edge_scatter_kernel<<<ew(Ed * Hd), 256, 0, stream>>>(ei, xin, pre, Ed);
    gin_combine_kernel<<<ew((int)NH), 256, 0, stream>>>(xin, pre, p[0], layer, (int)NH);
    gemm(pre, p[1] + (size_t)layer * Hd * Hd, p[2] + layer * Hd, h1, Nn, Hd, Hd, 0);
    bn_stats_kernel<<<Hd, 256, 0, stream>>>(h1, Nn, mean_, var_);
    bn_apply_relu_kernel<<<ew((int)NH), 256, 0, stream>>>(h1, (int)NH, mean_, var_,
                                                          p[3] + layer * Hd, p[4] + layer * Hd);
    gemm(h1, p[5] + (size_t)layer * Hd * Hd, p[6] + layer * Hd, xout, Nn, Hd, Hd, 0);
  };

  auto transformer = [&](float* X, int layer) {
    gemm(X, tf_qkv_w + (size_t)layer * 3 * Hd * Hd, tf_qkv_b + layer * 3 * Hd, qkv, NT, Hd, 3 * Hd, 0);
    attn_kernel<<<dim3(Bg * HEADS, Sq), 256, 0, stream>>>(qkv, ao);
    gemm(ao, tf_out_w + (size_t)layer * Hd * Hd, tf_out_b + layer * Hd, x1, NT, Hd, Hd, 0);
    add_ln_kernel<<<NT, Hd, 0, stream>>>(X, x1, tf_ln1_g + layer * Hd, tf_ln1_b + layer * Hd, x1);
    // FF1 with fused ReLU, f16 output straight into H16
    gemm(x1, tf_ff1_w + (size_t)layer * DFF * Hd, tf_ff1_b + layer * DFF,
         nullptr, NT, Hd, DFF, 1, H16);
    // FF2 consumes H16 without reconversion
    gemm_a16(H16, tf_ff2_w + (size_t)layer * Hd * DFF, tf_ff2_b + layer * Hd, ao, NT, DFF, Hd, 0);
    add_ln_kernel<<<NT, Hd, 0, stream>>>(x1, ao, tf_ln2_g + layer * Hd, tf_ln2_b + layer * Hd, X);
  };

  // embeddings
  gemm_emb(x_sc, emb_sc_w, emb_sc_b, sc);
  gemm_emb(x_fc, emb_fc_w, emb_fc_b, fcb);

  // layers 0..L-2 with fusion + BN/ReLU (faithful to source bug: sc takes fc half, fc keeps GIN output)
  for (int i = 0; i < 2; ++i) {
    gin(sc, sc, sc_ei, gsc, i);
    gin(fcb, fcb, fc_ei, gfc, i);
    concat_kernel<<<ew((int)TH), 256, 0, stream>>>(sc, fcb, fus, (int)TH);
    transformer(fus, i);
    extract_kernel<<<ew((int)NH), 256, 0, stream>>>(fus, sc, (int)NH);
    bn_stats_kernel<<<Hd, 256, 0, stream>>>(sc, Nn, mean_, var_);
    bn_apply_relu_kernel<<<ew((int)NH), 256, 0, stream>>>(sc, (int)NH, mean_, var_,
                                                          bns_g + i * Hd, bns_b + i * Hd);
    bn_stats_kernel<<<Hd, 256, 0, stream>>>(fcb, Nn, mean_, var_);
    bn_apply_relu_kernel<<<ew((int)NH), 256, 0, stream>>>(fcb, (int)NH, mean_, var_,
                                                          bns_g + i * Hd, bns_b + i * Hd);
  }

  // last GIN layer; fc result goes straight into its output slice
  gin(sc, sc, sc_ei, gsc, 2);
  gin(fcb, fc_out, fc_ei, gfc, 2);

  // final fusion (faithful source bug: reuses transformer layer index L-2 == 1)
  concat_kernel<<<ew((int)TH), 256, 0, stream>>>(sc, fc_out, fus, (int)TH);
  transformer(fus, 1);
  extract_kernel<<<ew((int)NH), 256, 0, stream>>>(fus, sc_out, (int)NH);

  // pooling + classifier
  pool_kernel<<<Bg, Hd, 0, stream>>>(sc_out, psc);
  pool_kernel<<<Bg, Hd, 0, stream>>>(fc_out, pfc);
  classifier_kernel<<<Bg, Hd, 0, stream>>>(psc, pfc, cls_w1, cls_b1, cls_w2, cls_b2, out);
}
